// MultiPathChebConv_58926951301257
// MI455X (gfx1250) — matrix operand order repaired and checked by reading, not compiled, not run on hardware
//
#include <hip/hip_runtime.h>
#include <hip/hip_bf16.h>

#define NROWS 4096
#define DIM   128
#define KMATS 3

typedef __attribute__((ext_vector_type(16))) __bf16 bf16x16;
typedef __attribute__((ext_vector_type(8)))  float  f32x8;

__device__ __forceinline__ unsigned int pack_bf16x2(float a, float b) {
  unsigned int ua = __float_as_uint(a);
  unsigned int ub = __float_as_uint(b);
  ua = (ua + 0x7FFFu + ((ua >> 16) & 1u)) >> 16;  // round-to-nearest-even
  ub = (ub + 0x7FFFu + ((ub >> 16) & 1u)) >> 16;
  return ua | (ub << 16);
}

// ---------------------------------------------------------------------------
// Stage 1: yT[p*3+k][col][m] = bf16( sum_d x[m][d] * W_p[k][d][col] )
// Stored transposed (col-major over m) so stage 2's B fragments are
// contiguous global_load_b128s in the same k order as the A frags.
// ---------------------------------------------------------------------------
__global__ __launch_bounds__(256) void yT_kernel(
    const float* __restrict__ x,
    const float* __restrict__ W0, const float* __restrict__ W1,
    const float* __restrict__ W2, const float* __restrict__ W3,
    unsigned short* __restrict__ yT) {
  int pk = blockIdx.y;               // 0..11
  int p  = pk / KMATS, k = pk % KMATS;
  const float* W = (p == 0 ? W0 : p == 1 ? W1 : p == 2 ? W2 : W3)
                 + (size_t)k * DIM * DIM;
  int idx = blockIdx.x * 256 + threadIdx.x;   // over 4096*128
  int col = idx & (DIM - 1);
  int m   = idx >> 7;
  const float* xr = x + (size_t)m * DIM;
  float acc = 0.0f;
#pragma unroll 4
  for (int d = 0; d < DIM; ++d)
    acc = fmaf(xr[d], W[(size_t)d * DIM + col], acc);
  unsigned int u = __float_as_uint(acc);
  u = (u + 0x7FFFu + ((u >> 16) & 1u)) >> 16;
  yT[(size_t)pk * DIM * NROWS + (size_t)col * NROWS + m] = (unsigned short)u;
}

// ---------------------------------------------------------------------------
// Stage 2 (dominant): out_p[n, :] = sum_k mats_p[k] @ yT[p,k] + b_p
// 32-row x 128-col slab per workgroup: 8 waves, TWO 16x16 C tiles per wave so
// each B fragment (L2-resident yT) feeds two v_wmma_f32_16x16x32_bf16.
// fp32 matrix stream -> bf16 in LDS (double buffered, one barrier per chunk).
// ---------------------------------------------------------------------------
__global__ __launch_bounds__(256) void big_kernel(
    const float* __restrict__ M0, const float* __restrict__ M1,
    const float* __restrict__ M2, const float* __restrict__ M3,
    const float* __restrict__ b0, const float* __restrict__ b1,
    const float* __restrict__ b2, const float* __restrict__ b3,
    const unsigned short* __restrict__ yT,
    float* __restrict__ outs) {
  const int path = blockIdx.y;
  const int n0   = blockIdx.x * 32;
  const float* mats = path == 0 ? M0 : path == 1 ? M1 : path == 2 ? M2 : M3;
  const float* bias = path == 0 ? b0 : path == 1 ? b1 : path == 2 ? b2 : b3;

  __shared__ __align__(16) unsigned short lds[2][32 * 128];   // 2 x 8KB bf16

  const int tid   = threadIdx.x;
  const int srow  = tid >> 3;          // staging: 32 rows, 8 threads/row
  const int smoff = (tid & 7) << 4;    // 16 floats per thread

  const int warp = tid >> 5;           // wave32: 8 waves, one 16-col tile each
  const int lane = tid & 31;
  const int nn   = lane & 15;
  const int half = lane >> 4;
  const int col0 = warp * 16;

  f32x8 acc0 = {};                     // rows n0 +  0..15
  f32x8 acc1 = {};                     // rows n0 + 16..31

  auto stage = [&](int c, int buf) {
    int k  = c >> 5;                   // 32 chunks of 128 per k
    int m0 = (c & 31) << 7;
    const float* src = mats + (size_t)k * NROWS * NROWS
                            + (size_t)(n0 + srow) * NROWS + m0 + smoff;
    float4 f0 = *(const float4*)src;
    float4 f1 = *(const float4*)(src + 4);
    float4 f2 = *(const float4*)(src + 8);
    float4 f3 = *(const float4*)(src + 12);
    uint4 s0, s1;
    s0.x = pack_bf16x2(f0.x, f0.y);
    s0.y = pack_bf16x2(f0.z, f0.w);
    s0.z = pack_bf16x2(f1.x, f1.y);
    s0.w = pack_bf16x2(f1.z, f1.w);
    s1.x = pack_bf16x2(f2.x, f2.y);
    s1.y = pack_bf16x2(f2.z, f2.w);
    s1.z = pack_bf16x2(f3.x, f3.y);
    s1.w = pack_bf16x2(f3.z, f3.w);
    unsigned short* dst = &lds[buf][srow * 128 + smoff];
    *(uint4*)dst       = s0;
    *(uint4*)(dst + 8) = s1;
  };

  const int NCH = (KMATS * NROWS) / 128;   // 96 chunks
  stage(0, 0);
  __syncthreads();
  for (int c = 0; c < NCH; ++c) {
    int b = c & 1;
    if (c + 1 < NCH) {
      stage(c + 1, b ^ 1);
      if (c + 2 < NCH) {               // global_prefetch_b8 for chunk c+2
        int k2 = (c + 2) >> 5;
        int m2 = ((c + 2) & 31) << 7;
        __builtin_prefetch(mats + (size_t)k2 * NROWS * NROWS
                                + (size_t)(n0 + srow) * NROWS + m2 + smoff, 0, 0);
      }
    }
    int k  = c >> 5;
    int m0 = (c & 31) << 7;
    const unsigned short* yTpk =
        yT + ((size_t)(path * KMATS + k) * DIM + (col0 + nn)) * NROWS;
#pragma unroll
    for (int kk = 0; kk < 4; ++kk) {   // 4 x (K=32) per 128-chunk
      union { uint4 u[2]; bf16x16 v; } A0, A1, B;
      // B 32x16 bf16 frag: lane = column, lanes<16 K=0..15, lanes>=16 K=16..31
      const unsigned short* bb = yTpk + m0 + kk * 32 + half * 16;
      B.u[0] = *(const uint4*)bb;
      B.u[1] = *(const uint4*)(bb + 8);
      // A 16x32 bf16 frags: lane<16 -> K 0..15, lane>=16 -> K 16..31
      const unsigned short* ab = &lds[b][nn * 128 + kk * 32 + half * 16];
      A0.u[0] = *(const uint4*)ab;
      A0.u[1] = *(const uint4*)(ab + 8);
      A1.u[0] = *(const uint4*)(ab + 16 * 128);
      A1.u[1] = *(const uint4*)(ab + 16 * 128 + 8);
      acc0 = __builtin_amdgcn_wmma_f32_16x16x32_bf16(
          false, A0.v, false, B.v, (short)0, acc0, false, false);
      acc1 = __builtin_amdgcn_wmma_f32_16x16x32_bf16(
          false, A1.v, false, B.v, (short)0, acc1, false, false);
    }
    __syncthreads();
  }
  // C/D layout: lanes 0-15: N=lane, M=v; lanes 16-31: N=lane-16, M=v+8
  float bv = bias[col0 + nn];
  float* op0 = outs + ((size_t)path * NROWS + n0 + half * 8) * DIM + col0 + nn;
  float* op1 = op0 + (size_t)16 * DIM;
#pragma unroll
  for (int v = 0; v < 8; ++v) {
    op0[(size_t)v * DIM] = acc0[v] + bv;
    op1[(size_t)v * DIM] = acc1[v] + bv;
  }
}

// ---------------------------------------------------------------------------
// Stage 3: gate = sigmoid([out_sym|out_fwd|out_bwd|out_corr] @ gate_W + gate_b)
//          result = 0.25 * (sum of paths) * gate
// ---------------------------------------------------------------------------
__global__ __launch_bounds__(256) void gate_kernel(
    const float* __restrict__ outs, const float* __restrict__ gW,
    const float* __restrict__ gb, float* __restrict__ out) {
  __shared__ float ap[16][512];
  const int n0  = blockIdx.x * 16;
  const int tid = threadIdx.x;
  for (int i = tid; i < 16 * 512; i += 256) {
    int r = i >> 9, j = i & 511;
    int path = j >> 7, col = j & 127;
    ap[r][j] = outs[((size_t)path * NROWS + n0 + r) * DIM + col];
  }
  __syncthreads();
  const int col = tid & 127;
  const int rg  = tid >> 7;            // 2 row-groups of 8
  float gbv = gb[col];
  float accv[8];
#pragma unroll
  for (int r = 0; r < 8; ++r) accv[r] = gbv;
  for (int j = 0; j < 512; ++j) {
    float w = gW[(size_t)j * DIM + col];
#pragma unroll
    for (int r = 0; r < 8; ++r)
      accv[r] = fmaf(ap[rg * 8 + r][j], w, accv[r]);
  }
#pragma unroll
  for (int r = 0; r < 8; ++r) {
    int row = rg * 8 + r;
    float s = ap[row][col] + ap[row][128 + col] +
              ap[row][256 + col] + ap[row][384 + col];
    float g = 1.0f / (1.0f + __expf(-accv[r]));
    out[(size_t)(n0 + row) * DIM + col] = 0.25f * s * g;
  }
}

extern "C" void kernel_launch(void* const* d_in, const int* in_sizes, int n_in,
                              void* d_out, int out_size, void* d_ws, size_t ws_size,
                              hipStream_t stream) {
  const float* x     = (const float*)d_in[0];
  const float* Msym  = (const float*)d_in[1];
  const float* Mfwd  = (const float*)d_in[2];
  const float* Mbwd  = (const float*)d_in[3];
  const float* Mcorr = (const float*)d_in[4];
  const float* Wsym  = (const float*)d_in[5];
  const float* bsym  = (const float*)d_in[6];
  const float* Wfwd  = (const float*)d_in[7];
  const float* bfwd  = (const float*)d_in[8];
  const float* Wbwd  = (const float*)d_in[9];
  const float* bbwd  = (const float*)d_in[10];
  const float* Wcorr = (const float*)d_in[11];
  const float* bcorr = (const float*)d_in[12];
  const float* gW    = (const float*)d_in[13];
  const float* gb    = (const float*)d_in[14];

  // workspace: yT bf16 [12][128][4096] (12 MB), then outs f32 [4][4096][128] (8 MB)
  unsigned short* yT = (unsigned short*)d_ws;
  float* outs = (float*)((char*)d_ws +
                         (size_t)4 * KMATS * DIM * NROWS * sizeof(unsigned short));

  dim3 gA((NROWS * DIM) / 256, 4 * KMATS);
  yT_kernel<<<gA, 256, 0, stream>>>(x, Wsym, Wfwd, Wbwd, Wcorr, yT);

  dim3 gB(NROWS / 32, 4);
  big_kernel<<<gB, 256, 0, stream>>>(Msym, Mfwd, Mbwd, Mcorr,
                                     bsym, bfwd, bbwd, bcorr, yT, outs);

  gate_kernel<<<NROWS / 16, 256, 0, stream>>>(outs, gW, gb, (float*)d_out);
}
